// MieScattering_27977416966530
// MI455X (gfx1250) — compile-verified, matrix operand order checked by reading, not run on hardware
//
#include <hip/hip_runtime.h>
#include <hip/hip_bf16.h>
#include <math.h>

#define NMAXO 135           // Mie truncation order (matches reference)
#define NBLK  9             // ceil(135/16) K-blocks of 16 orders (pad w/ zero weights)

// ---------------- complex float helpers (setup kernel only) ----------------
struct cplx { float r, i; };
__device__ __forceinline__ cplx cmulc(cplx a, cplx b){ return {a.r*b.r - a.i*b.i, a.r*b.i + a.i*b.r}; }
__device__ __forceinline__ cplx csubc(cplx a, cplx b){ return {a.r - b.r, a.i - b.i}; }
__device__ __forceinline__ cplx rscale(cplx a, float s){ return {a.r*s, a.i*s}; }
__device__ __forceinline__ cplx cinvc(cplx a){ float d = 1.0f/(a.r*a.r + a.i*a.i); return {a.r*d, -a.i*d}; }
__device__ __forceinline__ cplx cdivc(cplx a, cplx b){ return cmulc(a, cinvc(b)); }

__device__ __forceinline__ unsigned pk_bf16(float lo, float hi) {
  // truncate-to-bf16 and pack: lo -> [15:0] (even K = pi weight/value), hi -> [31:16] (odd K = tau)
  return (__float_as_uint(lo) >> 16) | (__float_as_uint(hi) & 0xffff0000u);
}

// ---------------------------------------------------------------------------
// Setup kernel (1 lane): compute an, bn via upward Riccati-Bessel recurrences
// (identical formulas to the reference), then pre-pack the WMMA B-matrices
// (32x16 bf16, 16 orders x {pi,tau} per K-block) into d_ws in the exact
// per-lane VGPR layout the wave32 WMMA expects.
// bw[b*256 + r*32 + lane] = B-tile block b, B-fragment VGPR r, lane.
// ---------------------------------------------------------------------------
__global__ void mie_setup_kernel(const float* __restrict__ wl,
                                 const float* __restrict__ rad,
                                 const float* __restrict__ mre,
                                 const float* __restrict__ mim,
                                 unsigned* __restrict__ bw) {
  if (threadIdx.x != 0 || blockIdx.x != 0) return;

  const float x  = 6.283185307179586f * rad[0] / (wl[0] * 1e-9f);
  const cplx  m  = {mre[0], mim[0]};
  const float inv = 1.0f / (x + 1e-12f);

  // real-argument psi/chi, upward
  float psi[NMAXO + 2], chi[NMAXO + 2];
  psi[0] = sinf(x);            psi[1] = psi[0]*inv - cosf(x);
  chi[0] = -cosf(x);           chi[1] = cosf(x)*inv - sinf(x);
  for (int i = 2; i < NMAXO + 2; ++i) {
    float coef = (2.0f*i - 1.0f) * inv;
    psi[i] = coef*psi[i-1] - psi[i-2];
    chi[i] = coef*chi[i-1] - chi[i-2];
  }

  // complex-argument psi for z = m*x
  cplx z    = {m.r * x, m.i * x};
  cplx invz = cinvc({z.r + 1e-12f, z.i});
  cplx sz   = {sinf(z.r)*coshf(z.i),  cosf(z.r)*sinhf(z.i)};
  cplx cz   = {cosf(z.r)*coshf(z.i), -sinf(z.r)*sinhf(z.i)};
  cplx psm[NMAXO + 2];
  psm[0] = sz;
  psm[1] = csubc(cmulc(psm[0], invz), cz);
  for (int i = 2; i < NMAXO + 2; ++i) {
    cplx coef = rscale(invz, 2.0f*i - 1.0f);
    psm[i] = csubc(cmulc(coef, psm[i-1]), psm[i-2]);
  }

  // weights f*(an,bn), n = 1..NMAXO
  float arw[NMAXO + 1], aiw[NMAXO + 1], brw[NMAXO + 1], biw[NMAXO + 1];
  for (int n = 1; n <= NMAXO; ++n) {
    float fn    = (float)n;
    float psip  = psi[n-1] - fn*inv*psi[n];
    cplx  xi_n  = {psi[n],   -chi[n]};
    cplx  xi_m1 = {psi[n-1], -chi[n-1]};
    cplx  xip   = csubc(xi_m1, rscale(xi_n, fn*inv));
    cplx  psmp  = csubc(psm[n-1], cmulc(rscale(invz, fn), psm[n]));
    cplx  mpsm  = cmulc(m, psm[n]);

    cplx num_a = csubc(rscale(mpsm, psip), rscale(psmp, psi[n]));
    cplx den_a = csubc(cmulc(mpsm, xip), cmulc(xi_n, psmp));
    den_a.r += 1e-30f;
    cplx an = cdivc(num_a, den_a);

    cplx num_b = csubc(rscale(psm[n], psip), cmulc(m, rscale(psmp, psi[n])));
    cplx den_b = csubc(cmulc(psm[n], xip), cmulc(m, cmulc(xi_n, psmp)));
    den_b.r += 1e-30f;
    cplx bn = cdivc(num_b, den_b);

    float f = (2.0f*fn + 1.0f) / (fn*(fn + 1.0f));
    arw[n] = f*an.r; aiw[n] = f*an.i; brw[n] = f*bn.r; biw[n] = f*bn.i;
  }

  // Pack B tiles: column c: 0->S1r(ar,br) 1->S1i(ai,bi) 2->S2r(br,ar) 3->S2i(bi,ai)
  // 16-bit 32x16 B layout (wave32): lane half h = l>>4, col c = l&15,
  // fragment VGPR r holds K = 2*j, 2*j+1 with j = (r&3) + 4*h + 8*(r>>2).
  for (int b = 0; b < NBLK; ++b)
    for (int r = 0; r < 8; ++r)
      for (int l = 0; l < 32; ++l) {
        int h = l >> 4, c = l & 15;
        int j = (r & 3) + 4*h + 8*(r >> 2);
        int n = 16*b + j + 1;
        float wpi = 0.0f, wtau = 0.0f;
        if (n <= NMAXO) {
          if      (c == 0) { wpi = arw[n]; wtau = brw[n]; }
          else if (c == 1) { wpi = aiw[n]; wtau = biw[n]; }
          else if (c == 2) { wpi = brw[n]; wtau = arw[n]; }
          else if (c == 3) { wpi = biw[n]; wtau = aiw[n]; }
        }
        bw[b*256 + r*32 + l] = pk_bf16(wpi, wtau);
      }
}

// ---------------------------------------------------------------------------
// Main kernel: each lane runs the pi/tau recurrence for its own angle (VALU),
// packs (pi,tau) pairs to bf16; ds_swizzle SWAPX16 exchanges K-slices between
// lane halves to build two 16x32 A tiles; v_wmma_f32_16x16x32_bf16 accumulates
// the four channel sums, co-executing with the VALU recurrence.
// ---------------------------------------------------------------------------
typedef __attribute__((ext_vector_type(16))) __bf16 v16bf;
typedef __attribute__((ext_vector_type(8)))  float  v8f;

union Frag { unsigned u[8]; v16bf v; };

__global__ __launch_bounds__(256) void mie_phase_kernel(
    const float* __restrict__ cos_theta,
    const float* __restrict__ wl,
    const float* __restrict__ rad,
    const unsigned* __restrict__ bw,
    float* __restrict__ out) {

  __shared__ float lds[8][2][16][17];   // [wave][tile][angle-row][channel] (+pad)

  const int tid  = threadIdx.x;
  const int lane = tid & 31;
  const int wave = tid >> 5;
  const int half = lane >> 4;
  const int angle = blockIdx.x * 256 + tid;

  const float mu = cos_theta[angle];
  const float x  = 6.283185307179586f * rad[0] / (wl[0] * 1e-9f);

  // Legendre pi/tau recurrence state (reference semantics: pi_1 = 3*mu, tau_1 = mu)
  float p1 = 3.0f * mu, p2 = 1.0f;

  v8f acc0 = {}, acc1 = {};
  unsigned pk[16];

#pragma unroll
  for (int b = 0; b < NBLK; ++b) {
    // 16 sequential recurrence steps (n is compile-time after full unroll)
#pragma unroll
    for (int j = 0; j < 16; ++j) {
      const int n = b*16 + j + 1;
      float piv, tav;
      if (n == 1) {
        piv = p1;           // 3*mu
        tav = mu;
      } else {
        const float fn = (float)n;
        const float c1 = (2.0f*fn + 1.0f) / fn;
        const float c2 = (fn + 1.0f) / fn;
        piv = c1*mu*p1 - c2*p2;
        tav = fn*mu*piv - (fn + 1.0f)*p1;
        p2 = p1; p1 = piv;
      }
      pk[j] = pk_bf16(piv, tav);
    }

    // Build A tiles + load pre-packed B tile
    Frag A0, A1, B;
#pragma unroll
    for (int r = 0; r < 8; ++r) {
      const int j0 = (r & 3) + 8*(r >> 2);   // K-chunk owned by half 0
      const unsigned own0 = pk[j0];
      const unsigned own1 = pk[j0 + 4];      // K-chunk owned by half 1
      // SWAPX16: lane l <-> lane l^16 (group-of-32, xor=0x10, and=0x1f)
      const unsigned s1 = (unsigned)__builtin_amdgcn_ds_swizzle((int)own1, 0x401f);
      const unsigned s0 = (unsigned)__builtin_amdgcn_ds_swizzle((int)own0, 0x401f);
      A0.u[r] = half ? s1 : own0;   // tile of angles base..base+15
      A1.u[r] = half ? own1 : s0;   // tile of angles base+16..base+31
      B.u[r]  = bw[b*256 + r*32 + lane];
    }

    acc0 = __builtin_amdgcn_wmma_f32_16x16x32_bf16(false, A0.v, false, B.v,
                                                   (short)0, acc0, false, false);
    acc1 = __builtin_amdgcn_wmma_f32_16x16x32_bf16(false, A1.v, false, B.v,
                                                   (short)0, acc1, false, false);
  }

  // C layout: lane = channel column (c = lane&15), VGPR r = angle row r + 8*half.
  const int col = lane & 15;
#pragma unroll
  for (int r = 0; r < 8; ++r) {
    const int row = r + 8*half;
    lds[wave][0][row][col] = acc0[r];
    lds[wave][1][row][col] = acc1[r];
  }
  __syncthreads();

  // Re-gather my angle's four channels and finish.
  const float s1r = lds[wave][half][col][0];
  const float s1i = lds[wave][half][col][1];
  const float s2r = lds[wave][half][col][2];
  const float s2i = lds[wave][half][col][3];
  const float inv2x2 = 0.5f / (x * x);
  out[angle] = (s1r*s1r + s1i*s1i + s2r*s2r + s2i*s2i) * inv2x2;
}

// ---------------------------------------------------------------------------
extern "C" void kernel_launch(void* const* d_in, const int* in_sizes, int n_in,
                              void* d_out, int out_size, void* d_ws, size_t ws_size,
                              hipStream_t stream) {
  const float* cos_theta = (const float*)d_in[0];
  const float* wavelength = (const float*)d_in[1];
  const float* radius     = (const float*)d_in[2];
  const float* m_real     = (const float*)d_in[3];
  const float* m_imag     = (const float*)d_in[4];
  float* out = (float*)d_out;
  unsigned* bw = (unsigned*)d_ws;          // NBLK*8*32 = 2304 dwords

  mie_setup_kernel<<<1, 32, 0, stream>>>(wavelength, radius, m_real, m_imag, bw);

  const int n = in_sizes[0];               // 1048576 angles
  const int blocks = n / 256;              // 256 threads = 8 waves, 32 angles/wave
  mie_phase_kernel<<<blocks, 256, 0, stream>>>(cos_theta, wavelength, radius, bw, out);
}
